// Compressor_63840393888338
// MI455X (gfx1250) — matrix operand order repaired and checked by reading, not compiled, not run on hardware
//
#include <hip/hip_runtime.h>

// MI455X (gfx1250) wave32 / WMMA implementation.
// Stage 1: kv_score = x @ W.T  (M=16384, N=384, K=7168), bf16 WMMA + f32 accum.
//   - Each block owns 64 M-rows x all 384 N-cols -> x (470 MB fp32) streamed
//     from HBM exactly once: ~20us floor at 23.3 TB/s. W (11 MB) is L2-resident.
//   - Double-buffered LDS W tiles (padded stride 40 bf16 -> conflict-free
//     ds_load_b128 banks), one barrier per k-step, software-pipelined A loads,
//     and a depth-3 register queue of B fragments so WMMAs wait on partial
//     dscnt instead of a full DS drain.
// Stage 2: softmax-over-CR compress + RMSNorm + RoPE + permutation scatter.
// d_ws holds the fp32 kv_score scratch: 16384*384*4 = 24 MB.

typedef __attribute__((ext_vector_type(16))) __bf16 v16bf;
typedef __attribute__((ext_vector_type(8)))  float  v8f;

constexpr int DIM      = 7168;   // K
constexpr int HD       = 192;
constexpr int N2       = 2 * HD; // 384 = N
constexpr int M_ROWS   = 16384;  // NUM_SEQS * SEQ_LEN
constexpr int TC       = 2048;   // compressed tokens per seq
constexpr int TOTAL_C  = 8192;
constexpr int TPB_SLOT = 64;
constexpr int BM       = 64;     // M rows per block
constexpr int KC       = 32;     // K per WMMA step
constexpr int SP       = 40;     // padded LDS row stride (bf16): banks = 4k mod 64

__device__ __forceinline__ __bf16 f2bf(float f) { return (__bf16)f; }

struct A4 { float4 a0, a1, a2, a3; };
struct B32 { uint4 lo, hi; };

__device__ __forceinline__ A4 load_a_raw(const float* __restrict__ x, long row,
                                         int kc, int half) {
  const float4* p = reinterpret_cast<const float4*>(x + row * DIM + kc + 8 * half);
  A4 r;
  r.a0 = p[0]; r.a1 = p[1];          // K run [8*half, +8)
  r.a2 = p[4]; r.a3 = p[5];          // K run [16+8*half, +8)
  return r;
}

__device__ __forceinline__ v16bf cvt_a(const A4& a) {
  v16bf f;
  f[0]=f2bf(a.a0.x);  f[1]=f2bf(a.a0.y);  f[2]=f2bf(a.a0.z);  f[3]=f2bf(a.a0.w);
  f[4]=f2bf(a.a1.x);  f[5]=f2bf(a.a1.y);  f[6]=f2bf(a.a1.z);  f[7]=f2bf(a.a1.w);
  f[8]=f2bf(a.a2.x);  f[9]=f2bf(a.a2.y);  f[10]=f2bf(a.a2.z); f[11]=f2bf(a.a2.w);
  f[12]=f2bf(a.a3.x); f[13]=f2bf(a.a3.y); f[14]=f2bf(a.a3.z); f[15]=f2bf(a.a3.w);
  return f;
}

__device__ __forceinline__ void load_w_tile(const float* __restrict__ W, int kc,
                                            __bf16* dst, int tid) {
#pragma unroll
  for (int it = 0; it < 12; ++it) {
    const int e  = tid * 4 + it * 1024;    // element in 384x32 logical tile
    const int n  = e >> 5;
    const int kk = e & 31;
    const float4 w4 = *reinterpret_cast<const float4*>(W + (long)n * DIM + kc + kk);
    __bf16* p = dst + n * SP + kk;
    p[0] = f2bf(w4.x); p[1] = f2bf(w4.y);
    p[2] = f2bf(w4.z); p[3] = f2bf(w4.w);
  }
}

__device__ __forceinline__ B32 load_b_raw(const __bf16* bbase, int n, int half) {
  // 16-byte-aligned pair of ds_load_b128 (padded stride breaks 32B alignment)
  const uint4* bp = reinterpret_cast<const uint4*>(bbase + n * SP + 8 * half);
  B32 r; r.lo = bp[0]; r.hi = bp[2];
  return r;
}

__global__ __launch_bounds__(256) void gemm_kv_bf16wmma(
    const float* __restrict__ x, const float* __restrict__ W,
    float* __restrict__ score) {
  __shared__ __bf16 sW[2][N2 * SP]; // 2 x 30 KB (double buffer, padded rows)

  const int tid  = threadIdx.x;
  const int wv   = tid >> 5;     // wave id 0..7
  const int lane = tid & 31;
  const int half = lane >> 4;    // 0 / 1
  const int mr   = lane & 15;
  const int mw   = wv >> 1;      // 0..3 -> M sub-tile
  const int nw   = wv & 1;       // 0..1 -> N half
  const int m0   = blockIdx.x * BM + mw * 16;
  const int n0   = nw * 192;
  const long row = (long)(m0 + mr);

  v8f acc[12];
#pragma unroll
  for (int i = 0; i < 12; ++i) acc[i] = (v8f){0.f,0.f,0.f,0.f,0.f,0.f,0.f,0.f};

  // ---- prologue: first W tile + first A chunk in flight ----
  A4 ar = load_a_raw(x, row, 0, half);
  load_w_tile(W, 0, &sW[0][0], tid);
  __syncthreads();

  constexpr int NIT = DIM / KC; // 224
  for (int ki = 0; ki < NIT; ++ki) {
    const int buf     = ki & 1;
    const int kc_next = (ki + 1) * KC;

    // Kick off next W tile into the other buffer (safe: it was last read in
    // iteration ki-1 and the trailing barrier of ki-1 has passed). Overlaps
    // with this iteration's WMMAs.
    if (ki + 1 < NIT) load_w_tile(W, kc_next, &sW[buf ^ 1][0], tid);

    const v16bf af = cvt_a(ar);

    // Kick off next A chunk loads (hide HBM latency behind the WMMAs).
    if (ki + 1 < NIT) {
      ar = load_a_raw(x, row, kc_next, half);
      __builtin_prefetch(x + row * DIM + kc_next + KC, 0, 3); // global_prefetch_b8
    }

    const __bf16* bbase = &sW[buf][0];

    // Depth-3 register queue of B fragments: DS returns in order, so each
    // WMMA only needs a partial s_wait_dscnt while 2 frags stay in flight.
    B32 bq[3];
    bq[0] = load_b_raw(bbase, n0 + 0 * 16 + mr, half);
    bq[1] = load_b_raw(bbase, n0 + 1 * 16 + mr, half);
    bq[2] = load_b_raw(bbase, n0 + 2 * 16 + mr, half);
#pragma unroll
    for (int nt = 0; nt < 12; ++nt) {
      const v16bf bf = __builtin_bit_cast(v16bf, bq[nt % 3]);
      if (nt + 3 < 12) bq[nt % 3] = load_b_raw(bbase, n0 + (nt + 3) * 16 + mr, half);
      acc[nt] = __builtin_amdgcn_wmma_f32_16x16x32_bf16(
          /*neg_a=*/false, af, /*neg_b=*/false, bf,
          /*c_mod=*/(short)0, acc[nt], /*reuse_a=*/false, /*reuse_b=*/false);
    }
    __syncthreads();
  }

  // ---- epilogue: C/D layout -> row m0 + v + 8*half, col n0 + nt*16 + mr ----
#pragma unroll
  for (int nt = 0; nt < 12; ++nt) {
    const int n = n0 + nt * 16 + mr;
#pragma unroll
    for (int v = 0; v < 8; ++v) {
      score[(long)(m0 + v + 8 * half) * N2 + n] = acc[nt][v];
    }
  }
}

// One wave32 per compressed token; 6 channels per lane (6*32 = 192 = HD).
__global__ __launch_bounds__(256) void kv_postprocess(
    const float* __restrict__ score, const float* __restrict__ ape,
    const float* __restrict__ norm_w, const float* __restrict__ cosb,
    const float* __restrict__ sinb, const int* __restrict__ pos_ids,
    const int* __restrict__ btab, float* __restrict__ kv_out,
    float* __restrict__ kv_cache) {
  const int wv   = threadIdx.x >> 5;
  const int lane = threadIdx.x & 31;
  const int t    = blockIdx.x * 8 + wv;

  const float* r0 = score + (long)(2 * t) * N2;
  const float* r1 = r0 + N2;

  float h[6];
  float ss = 0.f;
#pragma unroll
  for (int i = 0; i < 6; ++i) {
    const int j = lane + 32 * i;
    const float kv0 = r0[j],      kv1 = r1[j];
    const float g0  = r0[HD + j] + ape[j];
    const float g1  = r1[HD + j] + ape[HD + j];
    const float m   = fmaxf(g0, g1);
    const float e0  = __expf(g0 - m);
    const float e1  = __expf(g1 - m);
    const float c   = (e0 * kv0 + e1 * kv1) / (e0 + e1);
    h[i] = c;
    ss  += c * c;
  }
  // wave32 RMS reduction
#pragma unroll
  for (int off = 16; off > 0; off >>= 1) ss += __shfl_xor(ss, off, 32);
  const float inv = rsqrtf(ss * (1.0f / HD) + 1e-6f);
#pragma unroll
  for (int i = 0; i < 6; ++i) h[i] *= inv * norm_w[lane + 32 * i];

  // RoPE: r1 = channels [128,160), r2 = [160,192); ROPE/2 == 32 == warpSize.
  const int pos = pos_ids[t];
  const float c = cosb[pos * 32 + lane];
  const float s = sinb[pos * 32 + lane];
  const float o1 = h[4] * c - h[5] * s;
  const float o2 = h[5] * c + h[4] * s;

  const int seq    = t / TC;
  const int within = t % TC;
  const int slot   = btab[seq * (TC / TPB_SLOT) + within / TPB_SLOT] * TPB_SLOT
                   + (within % TPB_SLOT);

  float* o   = kv_out   + (long)t    * HD;
  float* cch = kv_cache + (long)slot * HD;
#pragma unroll
  for (int i = 0; i < 4; ++i) {
    o[lane + 32 * i]   = h[i];
    cch[lane + 32 * i] = h[i];
  }
  o[128 + lane] = o1; cch[128 + lane] = o1;
  o[160 + lane] = o2; cch[160 + lane] = o2;
}

extern "C" void kernel_launch(void* const* d_in, const int* in_sizes, int n_in,
                              void* d_out, int out_size, void* d_ws, size_t ws_size,
                              hipStream_t stream) {
  const float* x       = (const float*)d_in[0];
  const float* W       = (const float*)d_in[1];
  const float* ape     = (const float*)d_in[2];
  const float* norm_w  = (const float*)d_in[3];
  const float* cosb    = (const float*)d_in[4];
  const float* sinb    = (const float*)d_in[5];
  const int*   pos_ids = (const int*)d_in[6];
  const int*   btab    = (const int*)d_in[7];

  float* kv_out   = (float*)d_out;
  float* kv_cache = kv_out + (long)TOTAL_C * HD;
  float* score    = (float*)d_ws; // needs 16384*384*4 = 24 MB

  gemm_kv_bf16wmma<<<M_ROWS / BM, 256, 0, stream>>>(x, W, score);
  kv_postprocess<<<TOTAL_C / 8, 256, 0, stream>>>(score, ape, norm_w, cosb, sinb,
                                                  pos_ids, btab, kv_out, kv_cache);
}
